// PointNetSetAbstraction_42588895707400
// MI455X (gfx1250) — matrix operand order repaired and checked by reading, not compiled, not run on hardware
//
#include <hip/hip_runtime.h>
#include <hip/hip_bf16.h>

typedef _Float16 v16h __attribute__((ext_vector_type(16)));
typedef float    v8f  __attribute__((ext_vector_type(8)));

#define B_     8
#define N_     4096
#define NPOINT 1024
#define NSAMP  32
#define INCH   64
#define TDIM   128
#define LTOT   262144        // B * NPOINT * NSAMP
#define R2     0.04f

__device__ __forceinline__ float gelu_f(float x) {
    return 0.5f * x * (1.0f + erff(x * 0.70710678118654752440f));
}

// ---------------------------------------------------------------------------
// 1) Farthest point sampling: one block per batch, dist array in LDS.
// ---------------------------------------------------------------------------
__global__ void fps_kernel(const float* __restrict__ xyz,
                           float* __restrict__ newxyz_ws,
                           float* __restrict__ out_xyz) {
    __shared__ float dist[N_];
    __shared__ float rv[256];
    __shared__ int   ri[256];
    __shared__ float cc[3];
    const int b   = blockIdx.x;
    const int tid = threadIdx.x;
    const float* xb = xyz + (size_t)b * N_ * 3;

    for (int n = tid; n < N_; n += 256) dist[n] = 1e10f;
    int far = 0;
    __syncthreads();

    for (int i = 0; i < NPOINT; ++i) {
        if (tid == 0) {
            float cx = xb[far * 3 + 0], cy = xb[far * 3 + 1], cz = xb[far * 3 + 2];
            cc[0] = cx; cc[1] = cy; cc[2] = cz;
            size_t o = ((size_t)b * NPOINT + i) * 3;
            newxyz_ws[o + 0] = cx; newxyz_ws[o + 1] = cy; newxyz_ws[o + 2] = cz;
            out_xyz[o + 0] = cx;   out_xyz[o + 1] = cy;   out_xyz[o + 2] = cz;
        }
        __syncthreads();
        float cx = cc[0], cy = cc[1], cz = cc[2];
        float bv = -1.0f; int bi = 0;
        for (int n = tid; n < N_; n += 256) {
            float dx = xb[n * 3 + 0] - cx;
            float dy = xb[n * 3 + 1] - cy;
            float dz = xb[n * 3 + 2] - cz;
            float d  = dx * dx + dy * dy + dz * dz;
            float nd = fminf(dist[n], d);
            dist[n]  = nd;
            if (nd > bv) { bv = nd; bi = n; }
        }
        rv[tid] = bv; ri[tid] = bi;
        __syncthreads();
        for (int s = 128; s > 0; s >>= 1) {
            if (tid < s) {
                if (rv[tid + s] > rv[tid] ||
                    (rv[tid + s] == rv[tid] && ri[tid + s] < ri[tid])) {
                    rv[tid] = rv[tid + s]; ri[tid] = ri[tid + s];
                }
            }
            __syncthreads();
        }
        far = ri[0];
        __syncthreads();
    }
}

// ---------------------------------------------------------------------------
// 2) Ball query: one wave32 per centroid; ballot + prefix popcount compaction
// ---------------------------------------------------------------------------
__global__ void qbp_kernel(const float* __restrict__ xyz,
                           const float* __restrict__ newxyz,
                           int* __restrict__ idxg) {
    __shared__ int sidx[8][NSAMP];
    const int lane = threadIdx.x & 31;
    const int wv   = threadIdx.x >> 5;
    const int sg   = blockIdx.x * 8 + wv;          // 0 .. 8191
    const int b    = sg >> 10;
    const float* xb = xyz + (size_t)b * N_ * 3;
    const float* nz = newxyz + (size_t)sg * 3;
    const float cx = nz[0], cy = nz[1], cz = nz[2];

    int cnt = 0;
    for (int n0 = 0; n0 < N_ && cnt < NSAMP; n0 += 32) {
        int n = n0 + lane;
        float dx = xb[n * 3 + 0] - cx;
        float dy = xb[n * 3 + 1] - cy;
        float dz = xb[n * 3 + 2] - cz;
        bool inr = (dx * dx + dy * dy + dz * dz) <= R2;
        unsigned m = (unsigned)__ballot(inr);
        int pre = __popc(m & ((1u << lane) - 1u));
        if (inr && (cnt + pre) < NSAMP) sidx[wv][cnt + pre] = n;
        cnt += __popc(m);
    }
    if (cnt > NSAMP) cnt = NSAMP;
    __syncthreads();
    int first = (cnt > 0) ? sidx[wv][0] : 0;
    int v = (lane < cnt) ? sidx[wv][lane] : first;
    idxg[(size_t)sg * NSAMP + lane] = v;
}

// ---------------------------------------------------------------------------
// 3) Prep: temb (in LDS), per-batch fused biases, WMMA A-fragment packing
//    A-frag (16x32 f16): grp=kk>>4, r=kk&15, lane=mm+16*grp, half=r
// ---------------------------------------------------------------------------
__global__ void prep_kernel(const float* __restrict__ t_embed,
                            const float* __restrict__ tlw0, const float* __restrict__ tlb0,
                            const float* __restrict__ tlw1, const float* __restrict__ tlb1,
                            const float* __restrict__ cw0,  const float* __restrict__ cb0,
                            const float* __restrict__ cw1,  const float* __restrict__ cb1,
                            float* __restrict__ bias0, float* __restrict__ bias1,
                            _Float16* __restrict__ W0f, _Float16* __restrict__ W1f) {
    __shared__ float gt[B_ * TDIM];
    __shared__ float te0[B_ * 67];
    __shared__ float te1[B_ * 64];
    const int tid = threadIdx.x;

    for (int i = tid; i < B_ * TDIM; i += 256) gt[i] = gelu_f(t_embed[i]);
    __syncthreads();

    for (int i = tid; i < B_ * 67; i += 256) {
        int bb = i / 67, c = i % 67;
        float s = tlb0[c];
        const float* w = tlw0 + (size_t)c * TDIM;
        const float* g = gt + bb * TDIM;
        for (int d = 0; d < TDIM; ++d) s += g[d] * w[d];
        te0[i] = s;
    }
    for (int i = tid; i < B_ * 64; i += 256) {
        int bb = i >> 6, c = i & 63;
        float s = tlb1[c];
        const float* w = tlw1 + (size_t)c * TDIM;
        const float* g = gt + bb * TDIM;
        for (int d = 0; d < TDIM; ++d) s += g[d] * w[d];
        te1[i] = s;
    }
    __syncthreads();

    for (int i = tid; i < B_ * 64; i += 256) {      // bias0[b][o], O=64, K=67
        int bb = i >> 6, o = i & 63;
        float s = cb0[o];
        for (int c = 0; c < 67; ++c) s += cw0[o * 67 + c] * te0[bb * 67 + c];
        bias0[i] = s;
    }
    for (int i = tid; i < B_ * 128; i += 256) {     // bias1[b][o], O=128, K=64
        int bb = i >> 7, o = i & 127;
        float s = cb1[o];
        for (int c = 0; c < 64; ++c) s += cw1[o * 64 + c] * te1[bb * 64 + c];
        bias1[i] = s;
    }
    // W0: 64x67 padded to 64x96 -> 4 Mtiles x 3 Ksteps
    for (int i = tid; i < 64 * 96; i += 256) {
        int m = i / 96, k = i % 96;
        float v = (k < 67) ? cw0[m * 67 + k] : 0.0f;
        int mt = m >> 4, mm = m & 15, ks = k >> 5, kk = k & 31;
        int grp = kk >> 4, r = kk & 15;
        int lane = mm + (grp << 4);
        int half = r;
        W0f[((size_t)(mt * 3 + ks) * 32 + lane) * 16 + half] = (_Float16)v;
    }
    // W1: 128x64 -> 8 Mtiles x 2 Ksteps
    for (int i = tid; i < 128 * 64; i += 256) {
        int m = i >> 6, k = i & 63;
        float v = cw1[m * 64 + k];
        int mt = m >> 4, mm = m & 15, ks = k >> 5, kk = k & 31;
        int grp = kk >> 4, r = kk & 15;
        int lane = mm + (grp << 4);
        int half = r;
        W1f[((size_t)(mt * 2 + ks) * 32 + lane) * 16 + half] = (_Float16)v;
    }
}

// ---------------------------------------------------------------------------
// 4) Gather + concat -> B-fragment layout (32x16 f16: lane=n%16+16*(k/16),
//    half=k%16). One thread per output column, K padded 67 -> 96.
// ---------------------------------------------------------------------------
__global__ void pack_x0_kernel(const float* __restrict__ xyz,
                               const float* __restrict__ points,
                               const int* __restrict__ idxg,
                               const float* __restrict__ newxyz,
                               _Float16* __restrict__ X0f) {
    const int l = blockIdx.x * 256 + threadIdx.x;   // global column
    const int b = l >> 15, rem = l & 32767, s = rem >> 5, k = rem & 31;
    const int p = idxg[(((size_t)b << 10) + s) * NSAMP + k];
    const float* nz = newxyz + ((size_t)((b << 10) + s)) * 3;
    const float* pz = xyz + ((size_t)(b * N_ + p)) * 3;
    const int ntile = l >> 4, lcol = l & 15;
    _Float16* fb = X0f + (size_t)ntile * 3 * 512;   // 3 ksteps * 32 lanes * 16 halves
#define PUT0(c, v) { int ks=(c)>>5, kk=(c)&31; int lane=lcol+(((kk)>>4)<<4); \
                     fb[((size_t)ks*32+lane)*16 + (kk&15)] = (_Float16)(v); }
    PUT0(0, pz[0] - nz[0]);
    PUT0(1, pz[1] - nz[1]);
    PUT0(2, pz[2] - nz[2]);
    const float* pf = points + (size_t)b * INCH * N_ + p;
    for (int c = 0; c < INCH; ++c) PUT0(3 + c, pf[(size_t)c * N_]);
    for (int c = 67; c < 96; ++c)  PUT0(c, 0.0f);
#undef PUT0
}

// ---------------------------------------------------------------------------
// 5) WMMA GEMM: Y[m,col] = sum_k W[m,k] * X[k,col] + bias[col>>15][m]
//    8 waves/block, one 16-column tile per wave, MT output row-tiles.
// ---------------------------------------------------------------------------
template<int NK, int MT>
__global__ void gemm_wmma_kernel(const _Float16* __restrict__ Xfrag,
                                 const _Float16* __restrict__ Wfrag,
                                 const float* __restrict__ bias,
                                 float* __restrict__ Y) {
    constexpr int O = MT * 16;
    const int lane  = threadIdx.x & 31;
    const int wv    = threadIdx.x >> 5;
    const int ntile = blockIdx.x * 8 + wv;
    const size_t L  = LTOT;

    v16h bf[NK];
#pragma unroll
    for (int ks = 0; ks < NK; ++ks)
        bf[ks] = *(const v16h*)(Xfrag + (((size_t)ntile * NK + ks) * 32 + lane) * 16);

    const int col   = ntile * 16 + (lane & 15);
    const int bidx  = col >> 15;
    const int mbase = (lane >> 4) << 3;            // lanes 16..31 hold M = v+8

#pragma unroll
    for (int mt = 0; mt < MT; ++mt) {
        v8f acc = {};
#pragma unroll
        for (int ks = 0; ks < NK; ++ks) {
            v16h a = *(const v16h*)(Wfrag + (((size_t)mt * NK + ks) * 32 + lane) * 16);
            acc = __builtin_amdgcn_wmma_f32_16x16x32_f16(
                false, a, false, bf[ks], (short)0, acc, false, false);
        }
#pragma unroll
        for (int v = 0; v < 8; ++v) {
            int m = mt * 16 + v + mbase;
            Y[(size_t)m * L + col] = acc[v] + bias[bidx * O + m];
        }
    }
}

// ---------------------------------------------------------------------------
// 6) BN statistics: one block per channel -> mean, rsqrt(var+eps)
// ---------------------------------------------------------------------------
__global__ void bn_stats_kernel(const float* __restrict__ Y,
                                float* __restrict__ meanO,
                                float* __restrict__ rstdO) {
    __shared__ float as[256], aq[256];
    const int c = blockIdx.x, tid = threadIdx.x;
    const size_t L = LTOT;
    float s = 0.0f, ss = 0.0f;
    for (size_t l = tid; l < L; l += 256) {
        float v = Y[(size_t)c * L + l];
        s += v; ss += v * v;
    }
    as[tid] = s; aq[tid] = ss;
    __syncthreads();
    for (int st = 128; st > 0; st >>= 1) {
        if (tid < st) { as[tid] += as[tid + st]; aq[tid] += aq[tid + st]; }
        __syncthreads();
    }
    if (tid == 0) {
        float mean = as[0] / (float)L;
        float var  = aq[0] / (float)L - mean * mean;
        meanO[c] = mean;
        rstdO[c] = rsqrtf(var + 1e-5f);
    }
}

// ---------------------------------------------------------------------------
// 7) Apply BN + GELU to layer0 output, repack as layer1 B-fragments (K=64)
// ---------------------------------------------------------------------------
__global__ void apply_pack1_kernel(const float* __restrict__ Y1,
                                   const float* __restrict__ mean,
                                   const float* __restrict__ rstd,
                                   const float* __restrict__ g,
                                   const float* __restrict__ bt,
                                   _Float16* __restrict__ X1f) {
    const size_t l = blockIdx.x * 256 + threadIdx.x;
    const size_t L = LTOT;
    const int ntile = (int)(l >> 4), lcol = (int)(l & 15);
    for (int c = 0; c < 64; ++c) {
        float v = Y1[(size_t)c * L + l];
        float y = gelu_f((v - mean[c]) * rstd[c] * g[c] + bt[c]);
        int ks = c >> 5, kk = c & 31;
        int lane = lcol + ((kk >> 4) << 4);
        X1f[(((size_t)ntile * 2 + ks) * 32 + lane) * 16 + (kk & 15)] = (_Float16)y;
    }
}

// ---------------------------------------------------------------------------
// 8) BN + GELU + max over K=32 -> new_points (B,128,NPOINT)
// ---------------------------------------------------------------------------
__global__ void final_max_kernel(const float* __restrict__ Y2,
                                 const float* __restrict__ mean,
                                 const float* __restrict__ rstd,
                                 const float* __restrict__ g,
                                 const float* __restrict__ bt,
                                 float* __restrict__ out) {
    const int gid = blockIdx.x * 256 + threadIdx.x;   // (b, o, s)
    const int s = gid & 1023, o = (gid >> 10) & 127, b = gid >> 17;
    const size_t L = LTOT;
    const size_t base = (size_t)o * L + ((size_t)b << 15) + (size_t)s * NSAMP;
    const float mn = mean[o], rs = rstd[o], gg = g[o], bb = bt[o];
    float m = -3.4e38f;
    for (int k = 0; k < NSAMP; ++k) {
        float v = Y2[base + k];
        float y = gelu_f((v - mn) * rs * gg + bb);
        m = fmaxf(m, y);
    }
    out[B_ * NPOINT * 3 + ((size_t)(b * 128 + o)) * NPOINT + s] = m;
}

// ---------------------------------------------------------------------------
// Workspace layout (bytes)
// ---------------------------------------------------------------------------
#define OFF_NEWXYZ 0x0ULL          //  98304 B
#define OFF_IDX    0x20000ULL      //   1 MiB
#define OFF_BIAS0  0x122000ULL     //   2 KiB
#define OFF_BIAS1  0x123000ULL     //   4 KiB
#define OFF_STATS  0x124000ULL     // mean0/rstd0/mean1/rstd1
#define OFF_W0F    0x125000ULL     //  12 KiB
#define OFF_W1F    0x129000ULL     //  16 KiB
#define OFF_X0F    0x200000ULL     //  48 MiB
#define OFF_Y1     0x3200000ULL    //  64 MiB
#define OFF_X1F    0x7200000ULL    //  32 MiB
#define OFF_Y2     0x9200000ULL    // 128 MiB

extern "C" void kernel_launch(void* const* d_in, const int* in_sizes, int n_in,
                              void* d_out, int out_size, void* d_ws, size_t ws_size,
                              hipStream_t stream) {
    (void)in_sizes; (void)n_in; (void)out_size; (void)ws_size;
    const float* xyz   = (const float*)d_in[0];
    const float* pts   = (const float*)d_in[1];
    const float* temb  = (const float*)d_in[2];
    const float* cw0   = (const float*)d_in[3];
    const float* cb0   = (const float*)d_in[4];
    const float* tlw0  = (const float*)d_in[5];
    const float* tlb0  = (const float*)d_in[6];
    const float* bng0  = (const float*)d_in[7];
    const float* bnb0  = (const float*)d_in[8];
    const float* cw1   = (const float*)d_in[9];
    const float* cb1   = (const float*)d_in[10];
    const float* tlw1  = (const float*)d_in[11];
    const float* tlb1  = (const float*)d_in[12];
    const float* bng1  = (const float*)d_in[13];
    const float* bnb1  = (const float*)d_in[14];

    char* ws = (char*)d_ws;
    float*    newxyz = (float*)(ws + OFF_NEWXYZ);
    int*      idxg   = (int*)(ws + OFF_IDX);
    float*    bias0  = (float*)(ws + OFF_BIAS0);
    float*    bias1  = (float*)(ws + OFF_BIAS1);
    float*    mean0  = (float*)(ws + OFF_STATS);
    float*    rstd0  = mean0 + 64;
    float*    mean1  = rstd0 + 64;
    float*    rstd1  = mean1 + 128;
    _Float16* W0f    = (_Float16*)(ws + OFF_W0F);
    _Float16* W1f    = (_Float16*)(ws + OFF_W1F);
    _Float16* X0f    = (_Float16*)(ws + OFF_X0F);
    float*    Y1     = (float*)(ws + OFF_Y1);
    _Float16* X1f    = (_Float16*)(ws + OFF_X1F);
    float*    Y2     = (float*)(ws + OFF_Y2);
    float*    out    = (float*)d_out;

    fps_kernel<<<B_, 256, 0, stream>>>(xyz, newxyz, out);
    qbp_kernel<<<(B_ * NPOINT) / 8, 256, 0, stream>>>(xyz, newxyz, idxg);
    prep_kernel<<<1, 256, 0, stream>>>(temb, tlw0, tlb0, tlw1, tlb1,
                                       cw0, cb0, cw1, cb1,
                                       bias0, bias1, W0f, W1f);
    pack_x0_kernel<<<LTOT / 256, 256, 0, stream>>>(xyz, pts, idxg, newxyz, X0f);
    gemm_wmma_kernel<3, 4><<<(LTOT / 16) / 8, 256, 0, stream>>>(X0f, W0f, bias0, Y1);
    bn_stats_kernel<<<64, 256, 0, stream>>>(Y1, mean0, rstd0);
    apply_pack1_kernel<<<LTOT / 256, 256, 0, stream>>>(Y1, mean0, rstd0, bng0, bnb0, X1f);
    gemm_wmma_kernel<2, 8><<<(LTOT / 16) / 8, 256, 0, stream>>>(X1f, W1f, bias1, Y2);
    bn_stats_kernel<<<128, 256, 0, stream>>>(Y2, mean1, rstd1);
    final_max_kernel<<<(B_ * 128 * NPOINT) / 256, 256, 0, stream>>>(Y2, mean1, rstd1,
                                                                    bng1, bnb1, out);
}
